// DeformableConv2dBlock_26190710571194
// MI455X (gfx1250) — compile-verified, matrix operand order checked by reading, not run on hardware
//
#include <hip/hip_runtime.h>
#include <hip/hip_bf16.h>

// Deformable Conv2d block for gfx1250 (MI455X, wave32, WMMA).
// B=4, C=64, H=W=128, O=64, K=3, K2=9, stride=1, pad=1.
//
// Column layout is k-major: col j = k*64 + c  (k = ky*3+kx, c = channel), so the
// staging kernels write 128B contiguous per thread (global_store_b128 x8).
// Weight matrices are permuted to match at f16-conversion time.
//
// Pipeline:
//   1) convert_weights : -> Wom f16 [32 x 576], Wdef f16 [64 x 576] (k-major cols)
//   2) im2col_f16      : x -> Xcol f16 [65536 x 576]
//   3) gemm_offmask    : WMMA (A=pixels, B=Wom via LDS) -> offset f32 [65536x18], mask f32 [65536x9]
//   4) deform_sample   : bilinear sample + mask modulate -> Acol f16 (reuses Xcol buffer)
//   5) gemm_deform     : WMMA (A=Wdef via LDS, B=pixels) + bias -> out f32 NCHW, coalesced stores

#define Bn 4
#define Cn 64
#define Hn 128
#define Wn 128
#define On 64
#define K2n 9
#define KK 576          // C * K2
#define Mn 65536        // B * H * W
#define HWn 16384       // H * W
#define LDPAD 584       // LDS row stride in halfs: 1168B = 292 dwords = 4*73 -> conflict-free

typedef __attribute__((ext_vector_type(16))) _Float16 v16h;
typedef __attribute__((ext_vector_type(8)))  _Float16 v8h;
typedef __attribute__((ext_vector_type(8)))  float    v8f;

// ---------------------------------------------------------------------------
// Fragment loader for V_WMMA_F32_16X16X32_F16 operands (works for global or LDS).
// Row-major f16 source, row stride ldk halfs (1152B or 1168B rows: 16B aligned).
// CDNA5 16-bit A/B layout (CDNA5 ISA §7.12.2):
//   lanes 0-15  (row = lane)    : VGPR0-3 hold K=k0+0..7,  VGPR4-7 hold K=k0+16..23
//   lanes 16-31 (row = lane-16) : VGPR0-3 hold K=k0+8..15, VGPR4-7 hold K=k0+24..31
// ---------------------------------------------------------------------------
__device__ __forceinline__ v16h load_frag(const _Float16* base, int row0, int ldk,
                                          int k0, int lane) {
    int r    = lane & 15;
    int koff = (lane < 16) ? 0 : 8;
    const _Float16* p = base + (size_t)(row0 + r) * ldk + (k0 + koff);
    v8h lo = *(const v8h*)(p);        // K chunk [koff .. koff+7]
    v8h hi = *(const v8h*)(p + 16);   // K chunk [koff+16 .. koff+23]
    v16h out;
#pragma unroll
    for (int i = 0; i < 8; ++i) { out[i] = lo[i]; out[i + 8] = hi[i]; }
    return out;
}

// ---------------------------------------------------------------------------
// 1) Weight conversion to f16, permuted to k-major columns (col = k*64 + c).
//    Wom rows: 0..17 = offset_w, 18..26 = mask_w, 27..31 = zero pad.
// ---------------------------------------------------------------------------
__global__ void convert_weights(const float* __restrict__ offw,
                                const float* __restrict__ maskw,
                                const float* __restrict__ defw,
                                _Float16* __restrict__ Wom,
                                _Float16* __restrict__ Wdef) {
    int tid = blockIdx.x * blockDim.x + threadIdx.x;
    if (tid < 32 * KK) {
        int r = tid / KK, j = tid % KK;
        int k = j >> 6, c = j & 63;
        int src = c * K2n + k;                     // source col in [C,K,K] flat order
        float v = (r < 18) ? offw[r * KK + src]
                           : ((r < 27) ? maskw[(r - 18) * KK + src] : 0.0f);
        Wom[tid] = (_Float16)v;
    } else if (tid < 96 * KK) {
        int t = tid - 32 * KK;
        int o = t / KK, j = t % KK;
        int k = j >> 6, c = j & 63;
        Wdef[t] = (_Float16)defw[o * KK + c * K2n + k];
    }
}

// ---------------------------------------------------------------------------
// 2) f16 im2col. grid = (256, 9), block = 256. Thread = (m, k):
//    reads 64 channels (coalesced across lanes per c), writes 128B contiguous.
// ---------------------------------------------------------------------------
__global__ void im2col_f16(const float* __restrict__ x, _Float16* __restrict__ Xcol) {
    int m = blockIdx.x * blockDim.x + threadIdx.x;   // 0..65535
    int k = blockIdx.y;                              // 0..8
    int ky = k / 3, kx = k % 3;
    int b = m >> 14, hw = m & (HWn - 1);
    int h = hw >> 7, w = hw & (Wn - 1);
    int y = h + ky - 1, xw = w + kx - 1;
    bool inb = (y >= 0 && y < Hn && xw >= 0 && xw < Wn);
    const float* xp = x + ((size_t)b << 20) + (y << 7) + xw;   // + c*16384 per channel

    __attribute__((aligned(16))) _Float16 buf[Cn];
#pragma unroll
    for (int c = 0; c < Cn; ++c)
        buf[c] = (_Float16)(inb ? xp[c << 14] : 0.0f);

    v8h* dst = (v8h*)(Xcol + (size_t)m * KK + (k << 6));
#pragma unroll
    for (int i = 0; i < 8; ++i) dst[i] = ((const v8h*)buf)[i];
}

// ---------------------------------------------------------------------------
// 3) Offset+mask GEMM: [65536 x 576] x [576 x 32] via WMMA, f32 accumulate.
//    Wom staged in LDS (rows padded to 1168B -> conflict-free ds_load_b128).
//    grid = 512 x 256 (8 waves); wave -> one 16-row pixel tile, 2 channel tiles.
// ---------------------------------------------------------------------------
__global__ void __launch_bounds__(256)
gemm_offmask(const _Float16* __restrict__ Xcol, const _Float16* __restrict__ Wom,
             const float* __restrict__ offb, const float* __restrict__ maskb,
             float* __restrict__ offbuf, float* __restrict__ maskbuf) {
    __shared__ _Float16 sW[32 * LDPAD];
    int t = threadIdx.x;
#pragma unroll
    for (int i = 0; i < 9; ++i) {                  // 32 rows * 72 chunks = 2304 = 9*256
        int q = t + 256 * i;
        int r = q / 72, ck = q % 72;
        *(v8h*)(sW + r * LDPAD + ck * 8) = *(const v8h*)(Wom + r * KK + ck * 8);
    }
    __syncthreads();

    int wave = t >> 5, lane = t & 31;
    int row0 = (blockIdx.x * 8 + wave) * 16;

    v8f c0 = {}, c1 = {};
    for (int kk = 0; kk < KK; kk += 32) {
        v16h a  = load_frag(Xcol, row0, KK, kk, lane);
        v16h b0 = load_frag(sW,   0,  LDPAD, kk, lane);
        v16h b1 = load_frag(sW,   16, LDPAD, kk, lane);
        c0 = __builtin_amdgcn_wmma_f32_16x16x32_f16(false, a, false, b0, (short)0, c0, false, false);
        c1 = __builtin_amdgcn_wmma_f32_16x16x32_f16(false, a, false, b1, (short)0, c1, false, false);
    }

    // D layout: lanes 0-15 -> M(pixel) = r, N(ch) = lane; lanes 16-31 -> M = 8+r, N = lane-16.
    int n     = lane & 15;
    int mbase = row0 + ((lane < 16) ? 0 : 8);
#pragma unroll
    for (int r = 0; r < 8; ++r) {
        int m = mbase + r;
        offbuf[(size_t)m * 18 + n] = c0[r] + offb[n];        // channels 0..15: offsets
        int ch = 16 + n;                                      // channels 16..31
        if (ch < 18) {
            offbuf[(size_t)m * 18 + ch] = c1[r] + offb[ch];
        } else if (ch < 27) {
            float v = c1[r] + maskb[ch - 18];
            maskbuf[(size_t)m * 9 + (ch - 18)] = 1.0f / (1.0f + __expf(-v));
        }
    }
}

// ---------------------------------------------------------------------------
// 4) Deformable bilinear sampling + mask modulation -> Acol f16 (k-major cols).
//    grid = (256, 9). Thread = (m, k): 64-channel gather, 128B contiguous store.
// ---------------------------------------------------------------------------
__global__ void deform_sample(const float* __restrict__ x,
                              const float* __restrict__ offbuf,
                              const float* __restrict__ maskbuf,
                              _Float16* __restrict__ Acol) {
    int m = blockIdx.x * blockDim.x + threadIdx.x;   // 0..65535
    int k = blockIdx.y;                              // 0..8
    int b = m >> 14, hw = m & (HWn - 1);
    int h = hw >> 7, w = hw & (Wn - 1);
    int ky = k / 3, kx = k % 3;

    float dy = offbuf[(size_t)m * 18 + 2 * k];
    float dx = offbuf[(size_t)m * 18 + 2 * k + 1];
    float mv = maskbuf[(size_t)m * 9 + k];

    float py = dy + (float)(h + ky - 1);
    float px = dx + (float)(w + kx - 1);
    float y0f = floorf(py), x0f = floorf(px);
    int   y0 = (int)y0f, x0 = (int)x0f;
    int   y1 = y0 + 1,   x1 = x0 + 1;
    float wy1 = py - y0f, wx1 = px - x0f;
    float wy0 = 1.0f - wy1, wx0 = 1.0f - wx1;

    float vy0 = (y0 >= 0 && y0 < Hn) ? 1.0f : 0.0f;
    float vy1 = (y1 >= 0 && y1 < Hn) ? 1.0f : 0.0f;
    float vx0 = (x0 >= 0 && x0 < Wn) ? 1.0f : 0.0f;
    float vx1 = (x1 >= 0 && x1 < Wn) ? 1.0f : 0.0f;
    int y0c = min(max(y0, 0), Hn - 1), y1c = min(max(y1, 0), Hn - 1);
    int x0c = min(max(x0, 0), Wn - 1), x1c = min(max(x1, 0), Wn - 1);

    float w00 = wy0 * wx0 * vy0 * vx0, w01 = wy0 * wx1 * vy0 * vx1;
    float w10 = wy1 * wx0 * vy1 * vx0, w11 = wy1 * wx1 * vy1 * vx1;
    int i00 = (y0c << 7) + x0c, i01 = (y0c << 7) + x1c;
    int i10 = (y1c << 7) + x0c, i11 = (y1c << 7) + x1c;

    const float* xb = x + ((size_t)b << 20);          // b * C * H * W
    __attribute__((aligned(16))) _Float16 buf[Cn];
#pragma unroll 8
    for (int c = 0; c < Cn; ++c) {
        const float* xc = xb + ((size_t)c << 14);
        float v = w00 * xc[i00] + w01 * xc[i01] + w10 * xc[i10] + w11 * xc[i11];
        buf[c] = (_Float16)(v * mv);
    }
    v8h* dst = (v8h*)(Acol + (size_t)m * KK + (k << 6));
#pragma unroll
    for (int i = 0; i < 8; ++i) dst[i] = ((const v8h*)buf)[i];
}

// ---------------------------------------------------------------------------
// 5) Deform GEMM, operand-swapped: A = Wdef (M=o, from LDS), B = pixel tile
//    (N=pixels, from global). D lanes = consecutive pixels -> coalesced NCHW.
//    grid = 512 x 256 (8 waves); wave -> one 16-pixel tile x all 64 o.
// ---------------------------------------------------------------------------
__global__ void __launch_bounds__(256)
gemm_deform(const _Float16* __restrict__ Acol, const _Float16* __restrict__ Wdef,
            const float* __restrict__ defb, float* __restrict__ out) {
    __shared__ _Float16 sW[On * LDPAD];
    int t = threadIdx.x;
#pragma unroll
    for (int i = 0; i < 18; ++i) {                 // 64 rows * 72 chunks = 4608 = 18*256
        int q = t + 256 * i;
        int r = q / 72, ck = q % 72;
        *(v8h*)(sW + r * LDPAD + ck * 8) = *(const v8h*)(Wdef + r * KK + ck * 8);
    }
    __syncthreads();

    int wave = t >> 5, lane = t & 31;
    int row0 = (blockIdx.x * 8 + wave) * 16;       // pixel tile base

    v8f acc[4] = {{}, {}, {}, {}};
    for (int kk = 0; kk < KK; kk += 32) {
        v16h bfrag = load_frag(Acol, row0, KK, kk, lane);     // B operand: pixels
#pragma unroll
        for (int nt = 0; nt < 4; ++nt) {
            v16h afrag = load_frag(sW, nt * 16, LDPAD, kk, lane);  // A operand: Wdef
            acc[nt] = __builtin_amdgcn_wmma_f32_16x16x32_f16(false, afrag, false, bfrag,
                                                             (short)0, acc[nt], false, false);
        }
    }

    // D layout: lanes 0-15 -> M(o) = r, N(pix) = lane; lanes 16-31 -> M = 8+r, N = lane-16.
    int pix   = row0 + (lane & 15);
    int b     = pix >> 14, hw = pix & (HWn - 1);
    int obase = (lane < 16) ? 0 : 8;
#pragma unroll
    for (int nt = 0; nt < 4; ++nt) {
#pragma unroll
        for (int r = 0; r < 8; ++r) {
            int o = nt * 16 + obase + r;
            out[((size_t)(b * On + o) << 14) + hw] = acc[nt][r] + defb[o];
        }
    }
}

// ---------------------------------------------------------------------------
// Workspace layout (bytes):
//   Wom  f16 [32 x 576]        @ 0         ( 36,864 B)
//   Wdef f16 [64 x 576]        @ 36864     ( 73,728 B)
//   offbuf f32 [65536 x 18]    @ 110592    ( 4,718,592 B)
//   maskbuf f32 [65536 x 9]    @ 4829184   ( 2,359,296 B)
//   col f16 [65536 x 576]      @ 7188480   (75,497,472 B)  (Xcol, then Acol)
// total ~82.7 MB
// ---------------------------------------------------------------------------
extern "C" void kernel_launch(void* const* d_in, const int* in_sizes, int n_in,
                              void* d_out, int out_size, void* d_ws, size_t ws_size,
                              hipStream_t stream) {
    const float* x        = (const float*)d_in[0];
    const float* offset_w = (const float*)d_in[1];
    const float* offset_b = (const float*)d_in[2];
    const float* mask_w   = (const float*)d_in[3];
    const float* mask_b   = (const float*)d_in[4];
    const float* deform_w = (const float*)d_in[5];
    const float* deform_b = (const float*)d_in[6];
    float* out = (float*)d_out;

    char* ws = (char*)d_ws;
    _Float16* Wom     = (_Float16*)(ws + 0);
    _Float16* Wdef    = (_Float16*)(ws + 36864);
    float*    offbuf  = (float*)(ws + 110592);
    float*    maskbuf = (float*)(ws + 4829184);
    _Float16* col     = (_Float16*)(ws + 7188480);

    convert_weights<<<(96 * KK + 255) / 256, 256, 0, stream>>>(offset_w, mask_w, deform_w,
                                                               Wom, Wdef);

    dim3 gi(Mn / 256, K2n);
    im2col_f16<<<gi, 256, 0, stream>>>(x, col);

    gemm_offmask<<<Mn / (16 * 8), 256, 0, stream>>>(col, Wom, offset_b, mask_b,
                                                    offbuf, maskbuf);

    dim3 gs(Mn / 256, K2n);
    deform_sample<<<gs, 256, 0, stream>>>(x, offbuf, maskbuf, col);

    gemm_deform<<<Mn / (16 * 8), 256, 0, stream>>>(col, Wdef, deform_b, out);
}
